// OutProductMean_42949672960604
// MI455X (gfx1250) — compile-verified, matrix operand order checked by reading, not run on hardware
//
#include <hip/hip_runtime.h>
#include <hip/hip_bf16.h>

typedef __attribute__((ext_vector_type(16))) _Float16 v16h;
typedef __attribute__((ext_vector_type(8)))  _Float16 v8h;
typedef __attribute__((ext_vector_type(8)))  float    v8f;

static constexpr int   L_DIM = 512, F_DIM = 64, P_DIM = 32, Z_DIM = 128;
static constexpr float EPS_LN = 1e-10f, EPS_NORM = 0.001f;

// -------------------------------------------------------------------------
// Kernel 1: LayerNorm over F + dual projection (a -> f16, b -> f32)
// grid = 512 (one row), block = 64 (one thread per feature)
// -------------------------------------------------------------------------
__global__ __launch_bounds__(64) void ln_proj_kernel(
    const float* __restrict__ M,  const float* __restrict__ mask,
    const float* __restrict__ gm, const float* __restrict__ bt,
    const float* __restrict__ Wa, const float* __restrict__ ba,
    const float* __restrict__ Wb, const float* __restrict__ bb,
    _Float16* __restrict__ a16,   float* __restrict__ b32) {
  __shared__ float red[F_DIM];
  __shared__ float ysh[F_DIM];
  const int l = blockIdx.x;
  const int t = threadIdx.x;

  const float x = M[l * F_DIM + t];
  red[t] = x;
  __syncthreads();
  #pragma unroll
  for (int s = F_DIM / 2; s > 0; s >>= 1) {
    if (t < s) red[t] += red[t + s];
    __syncthreads();
  }
  const float mean = red[0] * (1.0f / F_DIM);
  __syncthreads();

  const float dc = x - mean;
  red[t] = dc * dc;
  __syncthreads();
  #pragma unroll
  for (int s = F_DIM / 2; s > 0; s >>= 1) {
    if (t < s) red[t] += red[t + s];
    __syncthreads();
  }
  const float var = red[0] * (1.0f / F_DIM);
  __syncthreads();

  // v_rsq_f32 approximation; error ~1e-6 rel, invisible under f16 rounding.
  ysh[t] = dc * __builtin_amdgcn_rsqf(var + EPS_LN) * gm[t] + bt[t];
  __syncthreads();

  const float mk = mask[l];
  if (t < P_DIM) {
    float s = 0.f;
    #pragma unroll
    for (int f = 0; f < F_DIM; f++) s += ysh[f] * Wa[t * F_DIM + f];
    a16[l * P_DIM + t] = (_Float16)((s + ba[t]) * mk);
  } else {
    const int p = t - P_DIM;
    float s = 0.f;
    #pragma unroll
    for (int f = 0; f < F_DIM; f++) s += ysh[f] * Wb[p * F_DIM + f];
    b32[l * P_DIM + p] = (s + bb[p]) * mk;
  }
}

// -------------------------------------------------------------------------
// Kernel 2: T[j, z, d] = sum_e Wz[z, d*32+e] * b[j, e]   (f32 acc -> f16)
// Note Wz[z, d*32+e] == Wz[(z*32+d)*32 + e], so rows are contiguous.
// grid = 512 (j), block = 256; each thread computes 16 dot-32s.
// -------------------------------------------------------------------------
__global__ __launch_bounds__(256) void t_kernel(
    const float* __restrict__ Wz, const float* __restrict__ b32,
    _Float16* __restrict__ T) {
  __shared__ float bsh[P_DIM];
  const int j = blockIdx.x;
  const int t = threadIdx.x;
  if (t < P_DIM) bsh[t] = b32[j * P_DIM + t];
  __syncthreads();

  #pragma unroll 4
  for (int r = 0; r < 16; r++) {
    const int idx = t + 256 * r;                 // idx = z*32 + d  (0..4095)
    const float* w = Wz + (size_t)idx * P_DIM;   // contiguous 128B run
    float s = 0.f;
    #pragma unroll
    for (int e = 0; e < P_DIM; e++) s += w[e] * bsh[e];
    T[(size_t)j * (Z_DIM * P_DIM) + idx] = (_Float16)s;
  }
}

// -------------------------------------------------------------------------
// Kernel 3: Z[i,j,z] = (sum_d a16[i,d]*T[j,z,d] + bz[z]) / (eps + m_i*m_j)
// One block per j (8 waves). Wave holds 8 B-frags (all z-tiles of T[j]),
// loops over 4 i-tiles: 32 v_wmma_f32_16x16x32_f16 per wave, NT stores.
//
// f16 fragment layout (CDNA5 ISA 7.12.2): lane 0-15 -> M/N row 0-15,
// per-lane v16h element e holds K = e + (e>=8 ? 8 : 0) + (lane>=16 ? 8 : 0),
// i.e. two contiguous 8-halfword (16B) runs at K-offsets {half*8, 16+half*8}.
//
// Epilogue note: scale the accumulators IN PLACE first, then store from the
// 8 distinct acc VGPRs. Funnelling the values through one data register
// forces an s_wait_xcnt between every store (WAR on the store-data reg);
// with distinct sources only one XCNT wait per tile is needed.
// -------------------------------------------------------------------------
__global__ __launch_bounds__(256) void z_kernel(
    const _Float16* __restrict__ a16, const _Float16* __restrict__ T,
    const float* __restrict__ mask,   const float* __restrict__ bz,
    float* __restrict__ Z) {
  const int j    = blockIdx.x;
  const int lane = threadIdx.x & 31;
  const int wv   = threadIdx.x >> 5;   // 0..7
  const int half = lane >> 4;          // 0/1
  const int ln   = lane & 15;

  // B fragments: B is K x N = d(32) x z(16); column n -> z = zt*16 + n.
  const _Float16* Tj = T + (size_t)j * (Z_DIM * P_DIM);
  v16h  Bf[8];
  float bzv[8];
  #pragma unroll
  for (int zt = 0; zt < 8; zt++) {
    const _Float16* p = Tj + (zt * 16 + ln) * P_DIM + half * 8;
    v8h lo = *(const v8h*)(p);
    v8h hi = *(const v8h*)(p + 16);
    Bf[zt] = __builtin_shufflevector(lo, hi, 0, 1, 2, 3, 4, 5, 6, 7,
                                             8, 9, 10, 11, 12, 13, 14, 15);
    bzv[zt] = bz[zt * 16 + ln];
  }
  const float mj = mask[j];

  #pragma unroll
  for (int k = 0; k < 4; k++) {
    const int i0 = (wv + 8 * k) * 16;
    // A fragment: A is M x K = i(16) x d(32), row m -> i = i0 + m.
    const _Float16* pa = a16 + (i0 + ln) * P_DIM + half * 8;
    v8h alo = *(const v8h*)(pa);
    v8h ahi = *(const v8h*)(pa + 16);
    v16h Af = __builtin_shufflevector(alo, ahi, 0, 1, 2, 3, 4, 5, 6, 7,
                                                8, 9, 10, 11, 12, 13, 14, 15);
    // D layout: VGPR r, lane -> (m = r + 8*half, n = ln); per-row scales via
    // hardware v_rcp_f32 (divisor is eps + 0/1, approx error ~1e-7 rel).
    float sc[8];
    #pragma unroll
    for (int r = 0; r < 8; r++) {
      const float mi = mask[i0 + r + 8 * half];
      sc[r] = __builtin_amdgcn_rcpf(EPS_NORM + mi * mj);
    }
    #pragma unroll
    for (int zt = 0; zt < 8; zt++) {
      v8f acc = {};
      acc = __builtin_amdgcn_wmma_f32_16x16x32_f16(
          /*neg_a=*/false, Af, /*neg_b=*/false, Bf[zt],
          /*c_mod=*/(short)0, acc, /*reuse_a=*/false, /*reuse_b=*/false);
      // Fused epilogue in place: 8 distinct result registers.
      #pragma unroll
      for (int r = 0; r < 8; r++) acc[r] = (acc[r] + bzv[zt]) * sc[r];
      float* dst0 = Z + ((size_t)(i0 + 8 * half) * L_DIM + j) * Z_DIM
                      + zt * 16 + ln;
      #pragma unroll
      for (int r = 0; r < 8; r++) {
        __builtin_nontemporal_store(acc[r], dst0 + (size_t)r * L_DIM * Z_DIM);
      }
    }
  }
}

// -------------------------------------------------------------------------
extern "C" void kernel_launch(void* const* d_in, const int* in_sizes, int n_in,
                              void* d_out, int out_size, void* d_ws, size_t ws_size,
                              hipStream_t stream) {
  const float* M     = (const float*)d_in[0];
  const float* Mmask = (const float*)d_in[1];
  const float* gamma = (const float*)d_in[2];
  const float* beta  = (const float*)d_in[3];
  const float* Wa    = (const float*)d_in[4];
  const float* ba    = (const float*)d_in[5];
  const float* Wb    = (const float*)d_in[6];
  const float* bb    = (const float*)d_in[7];
  const float* Wz    = (const float*)d_in[8];
  const float* bz    = (const float*)d_in[9];
  float* Z = (float*)d_out;

  char* ws = (char*)d_ws;
  _Float16* T   = (_Float16*)(ws);                                       // 4 MB
  _Float16* a16 = (_Float16*)(ws + (size_t)L_DIM * Z_DIM * P_DIM * 2);   // 32 KB
  float*    b32 = (float*)(ws + (size_t)L_DIM * Z_DIM * P_DIM * 2
                              + (size_t)L_DIM * P_DIM * 2);              // 64 KB

  hipLaunchKernelGGL(ln_proj_kernel, dim3(L_DIM), dim3(F_DIM), 0, stream,
                     M, Mmask, gamma, beta, Wa, ba, Wb, bb, a16, b32);
  hipLaunchKernelGGL(t_kernel, dim3(L_DIM), dim3(256), 0, stream, Wz, b32, T);
  hipLaunchKernelGGL(z_kernel, dim3(L_DIM), dim3(256), 0, stream,
                     a16, T, Mmask, bz, Z);
}
